// RNNInduce_80178449481859
// MI455X (gfx1250) — matrix operand order repaired and checked by reading, not compiled, not run on hardware
//
#include <hip/hip_runtime.h>
#include <hip/hip_bf16.h>

// ---------------------------------------------------------------------------
// RNNInduce on MI455X (gfx1250, wave32)
//   - bf16 WMMA (v_wmma_f32_16x16x32_bf16) for input-projection GEMMs + FC
//   - persistent-workgroup LSTM scans: W_hh split between VGPRs (bf16x8
//     vectors -> v_fma_mix_f32_bf16) and bank-tuned LDS; h broadcast via LDS
//   - global_prefetch_b8 lookahead on the xg stream
// ---------------------------------------------------------------------------

#define T_SEQ 32768
#define HID   256        // H
#define G4H   1024       // 4*H
#define DIN   64
#define D2H   512        // 2*H
#define NIP   1024       // inducing points
#define OUT2  64         // 2*OUT

#define WROW  132        // LDS weight row stride in dwords (16B aligned, bank-spread)

typedef __attribute__((ext_vector_type(16))) __bf16 v16bf;
typedef __attribute__((ext_vector_type(8)))  float  v8f;
typedef __attribute__((ext_vector_type(8)))  __bf16 bfv8;   // 16B = one b128

union FragBF { unsigned u[8]; v16bf v; };

__device__ __forceinline__ unsigned short f2bf(float f) {
    union { float f; unsigned u; } x; x.f = f;
    unsigned r = x.u + 0x7fffu + ((x.u >> 16) & 1u);   // round-to-nearest-even
    return (unsigned short)(r >> 16);
}
__device__ __forceinline__ float sigm(float x) { return 1.f / (1.f + __expf(-x)); }
__device__ __forceinline__ float tanh_fast(float x) {
    float e2 = __expf(2.f * x);
    return (e2 - 1.f) / (e2 + 1.f);
}

// ------------------------------ prep kernels -------------------------------

__global__ void cvt_bf16_kernel(const float* __restrict__ src,
                                unsigned short* __restrict__ dst, int n) {
    int i = blockIdx.x * blockDim.x + threadIdx.x;
    if (i < n) dst[i] = f2bf(src[i]);
}

__global__ void bias_sum_kernel(const float* __restrict__ a,
                                const float* __restrict__ b,
                                float* __restrict__ out, int n) {
    int i = blockIdx.x * blockDim.x + threadIdx.x;
    if (i < n) out[i] = a[i] + b[i];
}

// x = concat(Y[T,63], dT[T,1]) -> bf16 [T,64]
__global__ void pack_x_kernel(const float* __restrict__ Y,
                              const float* __restrict__ dT,
                              unsigned short* __restrict__ xbf) {
    int i = blockIdx.x * blockDim.x + threadIdx.x;
    if (i >= T_SEQ * DIN) return;
    int t = i >> 6, j = i & 63;
    float v = (j < 63) ? Y[t * 63 + j] : dT[t];
    xbf[i] = f2bf(v);
}

// fc_weight [512,64] fp32 -> transposed bf16 [64,512] to match W[N,K] layout
__global__ void fc_tcvt_kernel(const float* __restrict__ fc,
                               unsigned short* __restrict__ fcT) {
    int i = blockIdx.x * blockDim.x + threadIdx.x;
    if (i >= OUT2 * D2H) return;
    int nn = i / D2H, kk = i % D2H;
    fcT[i] = f2bf(fc[kk * OUT2 + nn]);
}

// -------------------- WMMA GEMM: C[M,N] = A[M,K] * W[N,K]^T + bias ----------
// A: bf16 row-major (lda = K), W: bf16 row-major [N,K], C: fp32 (ldc).
// grid.x = M/16, grid.y = N/64; 128 threads = 4 waves, one 16x16 tile/wave.

__global__ __launch_bounds__(128)
void gemm_wmma_bf16(const unsigned short* __restrict__ A, int lda,
                    const unsigned short* __restrict__ W, int K,
                    const float* __restrict__ bias,
                    float* __restrict__ C, int ldc) {
    const int wave = threadIdx.x >> 5;
    const int lane = threadIdx.x & 31;
    const int half = lane >> 4;
    const int lm   = lane & 15;
    const int m0   = blockIdx.x * 16;
    const int n0   = (blockIdx.y * 4 + wave) * 16;

    const unsigned short* arow = A + (size_t)(m0 + lm) * lda;
    const unsigned short* wrow = W + (size_t)(n0 + lm) * K;

    v8f acc = {};
    for (int k0 = 0; k0 < K; k0 += 32) {
        FragBF a, b;
#pragma unroll
        for (int v = 0; v < 8; ++v) {
            // A 16x32 bf16 layout: lane=(half,m); VGPR v -> K pair
            int ka = k0 + ((v < 4) ? 0 : 16) + half * 8 + (v & 3) * 2;
            a.u[v] = *(const unsigned*)(arow + ka);
            // B 32x16 bf16 layout: lane=(half,n); VGPR v -> K pair (same as A)
            int kb = k0 + ((v < 4) ? 0 : 16) + half * 8 + (v & 3) * 2;
            b.u[v] = *(const unsigned*)(wrow + kb);
        }
        acc = __builtin_amdgcn_wmma_f32_16x16x32_bf16(
            false, a.v, false, b.v, (short)0, acc, false, false);
    }
    const float bn = bias[n0 + lm];
#pragma unroll
    for (int r = 0; r < 8; ++r) {
        int row = m0 + r + half * 8;                 // C layout: VGPR r -> M
        C[(size_t)row * ldc + n0 + lm] = acc[r] + bn;
    }
}

// ------------------------- persistent LSTM scan ----------------------------
// One block per direction (grid.x = 2), 512 threads. Thread n owns gate rows
// n (i/f region, weights resident in 128 VGPRs as bf16x8 vectors) and n+512
// (g/o region, weights in LDS at 132-dword stride: 16B-aligned b128 reads,
// lane i starts at bank 4i -> each bank hit exactly twice per wave = minimum).
// h broadcast via LDS float4 reads; mixed-precision FMAs (v_fma_mix_f32_bf16).

__global__ __launch_bounds__(512)
void lstm_scan_kernel(const float* __restrict__ xg_f,
                      const float* __restrict__ xg_b,
                      const unsigned short* __restrict__ whh_f,
                      const unsigned short* __restrict__ whh_b,
                      unsigned short* __restrict__ hout /* bf16 [T,512] */) {
    const bool rev = (blockIdx.x == 1);
    const float* xg = rev ? xg_b : xg_f;
    const unsigned short* whh = rev ? whh_b : whh_f;
    const int colofs = rev ? HID : 0;
    const int n = threadIdx.x;

    __shared__ __align__(16) float h_lds[HID];
    __shared__ float gbuf[G4H];
    __shared__ __align__(16) unsigned wlds[512 * WROW];   // ~264 KB of 320 KB

    // Row n in VGPRs (fully unrolled -> stays in registers).
    bfv8 wreg[32];
    {
        const bfv8* wr = (const bfv8*)(whh + (size_t)n * HID);
#pragma unroll
        for (int q = 0; q < 32; ++q) wreg[q] = wr[q];
    }
    // Row n+512 staged into LDS.
    {
        const unsigned* src = (const unsigned*)(whh + (size_t)(n + 512) * HID);
        unsigned* dst = wlds + n * WROW;
        for (int p = 0; p < 128; ++p) dst[p] = src[p];
    }

    if (n < HID) h_lds[n] = 0.f;
    float c = 0.f;
    __syncthreads();

    const float4* h4 = (const float4*)h_lds;
    const bfv8*   wl = (const bfv8*)(wlds + n * WROW);

    for (int s = 0; s < T_SEQ; ++s) {
        const int t = rev ? (T_SEQ - 1 - s) : s;
        if (s + 8 < T_SEQ) {                       // global_prefetch_b8
            int tp = rev ? (T_SEQ - 9 - s) : (s + 8);
            __builtin_prefetch(xg + (size_t)tp * G4H + n, 0, 0);
        }
        float a0 = xg[(size_t)t * G4H + n];
        float a1 = xg[(size_t)t * G4H + n + 512];
#pragma unroll
        for (int q = 0; q < 32; ++q) {
            float4 hva = h4[2 * q];                // LDS broadcast (conflict-free)
            float4 hvb = h4[2 * q + 1];
            bfv8 w0 = wreg[q];
            bfv8 w1 = wl[q];                       // ds_load_b128, bank-tuned
            a0 = fmaf((float)w0[0], hva.x, a0);
            a0 = fmaf((float)w0[1], hva.y, a0);
            a0 = fmaf((float)w0[2], hva.z, a0);
            a0 = fmaf((float)w0[3], hva.w, a0);
            a0 = fmaf((float)w0[4], hvb.x, a0);
            a0 = fmaf((float)w0[5], hvb.y, a0);
            a0 = fmaf((float)w0[6], hvb.z, a0);
            a0 = fmaf((float)w0[7], hvb.w, a0);
            a1 = fmaf((float)w1[0], hva.x, a1);
            a1 = fmaf((float)w1[1], hva.y, a1);
            a1 = fmaf((float)w1[2], hva.z, a1);
            a1 = fmaf((float)w1[3], hva.w, a1);
            a1 = fmaf((float)w1[4], hvb.x, a1);
            a1 = fmaf((float)w1[5], hvb.y, a1);
            a1 = fmaf((float)w1[6], hvb.z, a1);
            a1 = fmaf((float)w1[7], hvb.w, a1);
        }
        gbuf[n]       = a0;
        gbuf[n + 512] = a1;
        __syncthreads();
        if (n < HID) {                             // gate combine, order i,f,g,o
            float gi = gbuf[n];
            float gf = gbuf[HID + n];
            float gg = gbuf[2 * HID + n];
            float go = gbuf[3 * HID + n];
            c = sigm(gf) * c + sigm(gi) * tanh_fast(gg);
            float h = sigm(go) * tanh_fast(c);
            h_lds[n] = h;
            hout[(size_t)t * D2H + colofs + n] = f2bf(h);
        }
        __syncthreads();
    }
}

// --------------------- FC + gather via WMMA (NI x 512 x 64) ----------------
// grid.x = NI/16 = 64 blocks; 4 waves -> 4 n-tiles. A rows gathered by idx.

__global__ __launch_bounds__(128)
void fc_gather_wmma(const unsigned short* __restrict__ h1,
                    const unsigned short* __restrict__ fcT, /* [64,512] bf16 */
                    const int* __restrict__ idx,
                    float* __restrict__ out) {
    const int wave = threadIdx.x >> 5;
    const int lane = threadIdx.x & 31;
    const int half = lane >> 4;
    const int lm   = lane & 15;
    const int m0   = blockIdx.x * 16;
    const int n0   = wave * 16;

    const int arow = idx[m0 + lm];
    const unsigned short* ap = h1 + (size_t)arow * D2H;
    const unsigned short* wp = fcT + (size_t)(n0 + lm) * D2H;

    v8f acc = {};
    for (int k0 = 0; k0 < D2H; k0 += 32) {
        FragBF a, b;
#pragma unroll
        for (int v = 0; v < 8; ++v) {
            int ka = k0 + ((v < 4) ? 0 : 16) + half * 8 + (v & 3) * 2;
            a.u[v] = *(const unsigned*)(ap + ka);
            int kb = k0 + ((v < 4) ? 0 : 16) + half * 8 + (v & 3) * 2;
            b.u[v] = *(const unsigned*)(wp + kb);
        }
        acc = __builtin_amdgcn_wmma_f32_16x16x32_bf16(
            false, a.v, false, b.v, (short)0, acc, false, false);
    }
#pragma unroll
    for (int r = 0; r < 8; ++r) {
        int m   = m0 + r + half * 8;
        int col = n0 + lm;
        float v = acc[r];
        if (col < 32) out[(size_t)m * 32 + col] = v;                         // sel[:, :32]
        else          out[(size_t)NIP * 32 + (size_t)m * 32 + (col - 32)] = v; // sel[:, 32:]
    }
}

// ---------------------------------------------------------------------------

extern "C" void kernel_launch(void* const* d_in, const int* in_sizes, int n_in,
                              void* d_out, int out_size, void* d_ws, size_t ws_size,
                              hipStream_t stream) {
    const float* Y         = (const float*)d_in[0];
    const float* dT        = (const float*)d_in[1];
    const int*   induce    = (const int*)  d_in[2];
    const float* wih0f     = (const float*)d_in[3];
    const float* whh0f     = (const float*)d_in[4];
    const float* bih0f     = (const float*)d_in[5];
    const float* bhh0f     = (const float*)d_in[6];
    const float* wih0b     = (const float*)d_in[7];
    const float* whh0b     = (const float*)d_in[8];
    const float* bih0b     = (const float*)d_in[9];
    const float* bhh0b     = (const float*)d_in[10];
    const float* wih1f     = (const float*)d_in[11];
    const float* whh1f     = (const float*)d_in[12];
    const float* bih1f     = (const float*)d_in[13];
    const float* bhh1f     = (const float*)d_in[14];
    const float* wih1b     = (const float*)d_in[15];
    const float* whh1b     = (const float*)d_in[16];
    const float* bih1b     = (const float*)d_in[17];
    const float* bhh1b     = (const float*)d_in[18];
    const float* fcw       = (const float*)d_in[19];
    float* out             = (float*)d_out;

    // ---------------- workspace carve-out (aligned 256B) ----------------
    char* ws = (char*)d_ws;
    size_t off = 0;
    auto carve = [&](size_t bytes) -> char* {
        char* p = ws + off;
        off = (off + bytes + 255) & ~(size_t)255;
        return p;
    };
    unsigned short* xbf      = (unsigned short*)carve((size_t)T_SEQ * DIN * 2);
    unsigned short* wih0f_b  = (unsigned short*)carve((size_t)G4H * DIN * 2);
    unsigned short* wih0b_b  = (unsigned short*)carve((size_t)G4H * DIN * 2);
    unsigned short* wih1f_b  = (unsigned short*)carve((size_t)G4H * D2H * 2);
    unsigned short* wih1b_b  = (unsigned short*)carve((size_t)G4H * D2H * 2);
    unsigned short* whh0f_b  = (unsigned short*)carve((size_t)G4H * HID * 2);
    unsigned short* whh0b_b  = (unsigned short*)carve((size_t)G4H * HID * 2);
    unsigned short* whh1f_b  = (unsigned short*)carve((size_t)G4H * HID * 2);
    unsigned short* whh1b_b  = (unsigned short*)carve((size_t)G4H * HID * 2);
    unsigned short* fcT_b    = (unsigned short*)carve((size_t)OUT2 * D2H * 2);
    float* bias0f            = (float*)carve((size_t)G4H * 4);
    float* bias0b            = (float*)carve((size_t)G4H * 4);
    float* bias1f            = (float*)carve((size_t)G4H * 4);
    float* bias1b            = (float*)carve((size_t)G4H * 4);
    unsigned short* h0       = (unsigned short*)carve((size_t)T_SEQ * D2H * 2);
    unsigned short* h1       = (unsigned short*)carve((size_t)T_SEQ * D2H * 2);
    float* xgf               = (float*)carve((size_t)T_SEQ * G4H * 4); // reused L0/L1
    float* xgb               = (float*)carve((size_t)T_SEQ * G4H * 4); // reused L0/L1
    (void)ws_size; (void)n_in; (void)in_sizes; (void)out_size;

    const int TB = 256;
    auto blocks = [&](size_t n) { return (unsigned)((n + TB - 1) / TB); };

    // ---------------- prep: pack x, convert weights, sum biases ----------
    pack_x_kernel<<<blocks((size_t)T_SEQ * DIN), TB, 0, stream>>>(Y, dT, xbf);
    cvt_bf16_kernel<<<blocks(G4H * DIN), TB, 0, stream>>>(wih0f, wih0f_b, G4H * DIN);
    cvt_bf16_kernel<<<blocks(G4H * DIN), TB, 0, stream>>>(wih0b, wih0b_b, G4H * DIN);
    cvt_bf16_kernel<<<blocks((size_t)G4H * D2H), TB, 0, stream>>>(wih1f, wih1f_b, G4H * D2H);
    cvt_bf16_kernel<<<blocks((size_t)G4H * D2H), TB, 0, stream>>>(wih1b, wih1b_b, G4H * D2H);
    cvt_bf16_kernel<<<blocks((size_t)G4H * HID), TB, 0, stream>>>(whh0f, whh0f_b, G4H * HID);
    cvt_bf16_kernel<<<blocks((size_t)G4H * HID), TB, 0, stream>>>(whh0b, whh0b_b, G4H * HID);
    cvt_bf16_kernel<<<blocks((size_t)G4H * HID), TB, 0, stream>>>(whh1f, whh1f_b, G4H * HID);
    cvt_bf16_kernel<<<blocks((size_t)G4H * HID), TB, 0, stream>>>(whh1b, whh1b_b, G4H * HID);
    fc_tcvt_kernel<<<blocks(OUT2 * D2H), TB, 0, stream>>>(fcw, fcT_b);
    bias_sum_kernel<<<blocks(G4H), TB, 0, stream>>>(bih0f, bhh0f, bias0f, G4H);
    bias_sum_kernel<<<blocks(G4H), TB, 0, stream>>>(bih0b, bhh0b, bias0b, G4H);
    bias_sum_kernel<<<blocks(G4H), TB, 0, stream>>>(bih1f, bhh1f, bias1f, G4H);
    bias_sum_kernel<<<blocks(G4H), TB, 0, stream>>>(bih1b, bhh1b, bias1b, G4H);

    // ---------------- layer 0: xg = x @ Wih^T + b  (WMMA) -----------------
    dim3 gg(T_SEQ / 16, G4H / 64);
    gemm_wmma_bf16<<<gg, 128, 0, stream>>>(xbf, DIN, wih0f_b, DIN, bias0f, xgf, G4H);
    gemm_wmma_bf16<<<gg, 128, 0, stream>>>(xbf, DIN, wih0b_b, DIN, bias0b, xgb, G4H);

    // ---------------- layer 0 scans (fwd + bwd on separate WGPs) ----------
    lstm_scan_kernel<<<2, 512, 0, stream>>>(xgf, xgb, whh0f_b, whh0b_b, h0);

    // ---------------- layer 1: xg = h0 @ Wih^T + b  (reuse xg bufs) -------
    gemm_wmma_bf16<<<gg, 128, 0, stream>>>(h0, D2H, wih1f_b, D2H, bias1f, xgf, G4H);
    gemm_wmma_bf16<<<gg, 128, 0, stream>>>(h0, D2H, wih1b_b, D2H, bias1b, xgb, G4H);

    // ---------------- layer 1 scans --------------------------------------
    lstm_scan_kernel<<<2, 512, 0, stream>>>(xgf, xgb, whh1f_b, whh1b_b, h1);

    // ---------------- FC on gathered induce rows (WMMA) -------------------
    fc_gather_wmma<<<NIP / 16, 128, 0, stream>>>(h1, fcT_b, induce, out);
}
